// LongformerClass_36601711297104
// MI455X (gfx1250) — matrix operand order repaired and checked by reading, not compiled, not run on hardware
//
#include <hip/hip_runtime.h>
#include <hip/hip_bf16.h>
#include <math.h>

typedef __bf16 bf16;
typedef __attribute__((ext_vector_type(16))) __bf16 v16bf;
typedef __attribute__((ext_vector_type(8)))  __bf16 v8bf;
typedef __attribute__((ext_vector_type(4)))  __bf16 v4bf;
typedef __attribute__((ext_vector_type(8)))  float  v8f;
typedef __attribute__((ext_vector_type(4)))  float  v4f;

#define NEGV (-1.0e9f)

// ---------------------------------------------------------------------------
// WMMA helpers (CDNA5 gfx1250, wave32). D = A(16x32 bf16) * B(32x16 bf16) + C(f32)
// ---------------------------------------------------------------------------
__device__ __forceinline__ v8f wmma_bf16(v16bf a, v16bf b, v8f c) {
  return __builtin_amdgcn_wmma_f32_16x16x32_bf16(false, a, false, b, (short)0, c,
                                                 false, false);
}

// A fragment from LDS, tile stored row-major [16+][ld] (M rows, K contiguous).
// ISA: lanes 0-15 -> M=lane, K={0..7,16..23}; lanes 16-31 -> K={8..15,24..31}
__device__ __forceinline__ v16bf load_a_lds(const bf16* base, int ld, int lane) {
  int m  = lane & 15;
  int kb = (lane & 16) ? 8 : 0;
  const bf16* p = base + m * ld;
  v8bf lo = *(const v8bf*)(p + kb);
  v8bf hi = *(const v8bf*)(p + 16 + kb);
  v16bf f;
#pragma unroll
  for (int e = 0; e < 8; ++e) { f[e] = lo[e]; f[8 + e] = hi[e]; }
  return f;
}

// B fragment from LDS, tile stored N-major: element (K=k, N=n) at base[n*ld + k].
// ISA: lanes 0-15 -> N=lane, K={0..7,16..23}; lanes 16-31 -> N=lane-16, K={8..15,24..31}
__device__ __forceinline__ v16bf load_b_lds(const bf16* base, int ld, int lane) {
  int n  = lane & 15;
  int kb = (lane & 16) ? 8 : 0;
  const bf16* p = base + n * ld + kb;
  v8bf lo = *(const v8bf*)(p);
  v8bf hi = *(const v8bf*)(p + 16);
  v16bf f;
#pragma unroll
  for (int e = 0; e < 8; ++e) { f[e] = lo[e]; f[8 + e] = hi[e]; }
  return f;
}

// A fragment straight from global fp32 (vectorized, scaled, cvt to bf16 in regs)
__device__ __forceinline__ v16bf load_a_gf32(const float* base, int ld, int lane,
                                             float scale) {
  int m  = lane & 15;
  int kb = (lane & 16) ? 8 : 0;
  const float* p = base + (size_t)m * ld;
  v4f x0 = *(const v4f*)(p + kb);
  v4f x1 = *(const v4f*)(p + kb + 4);
  v4f x2 = *(const v4f*)(p + 16 + kb);
  v4f x3 = *(const v4f*)(p + 16 + kb + 4);
  v16bf f;
#pragma unroll
  for (int e = 0; e < 4; ++e) {
    f[e]      = (bf16)(x0[e] * scale);
    f[4 + e]  = (bf16)(x1[e] * scale);
    f[8 + e]  = (bf16)(x2[e] * scale);
    f[12 + e] = (bf16)(x3[e] * scale);
  }
  return f;
}

// ---------------------------------------------------------------------------
// Generic WMMA GEMM: C[M,N] = act(A[M,K] @ W[K,N] + bias [+ res])
// block tile 128x128, BK=64, 256 threads = 8 waves, each wave a 32x64 tile.
// mode: 0 = none, 1 = exact GELU, 2 = residual add
// ---------------------------------------------------------------------------
__global__ __launch_bounds__(256) void gemm_wmma(
    const float* __restrict__ A, const float* __restrict__ W,
    const float* __restrict__ bias, const float* __restrict__ res,
    float* __restrict__ C, int M, int N, int K, int mode) {
  __shared__ __align__(16) bf16 As[128 * 64];   // [m][k]
  __shared__ __align__(16) bf16 Bs[128 * 64];   // [n][k]
  const int tid = threadIdx.x, lane = tid & 31, wid = tid >> 5;
  const int m0 = blockIdx.y * 128, n0 = blockIdx.x * 128;
  const int wm = (wid & 3) * 32, wn = (wid >> 2) * 64;
  v8f acc[2][4] = {};

  for (int k0 = 0; k0 < K; k0 += 64) {
    // stage A (128x64): float4 global loads, packed bf16x4 LDS stores
#pragma unroll
    for (int ci = tid; ci < 2048; ci += 256) {
      int m = ci >> 4, k = (ci & 15) * 4;
      v4f x = *(const v4f*)&A[(size_t)(m0 + m) * K + k0 + k];
      v4bf w;
#pragma unroll
      for (int j = 0; j < 4; ++j) w[j] = (bf16)x[j];
      *(v4bf*)&As[m * 64 + k] = w;
    }
    // stage B transposed (Bs[n][k]): float4 global loads along N
#pragma unroll
    for (int ci = tid; ci < 2048; ci += 256) {
      int k = ci >> 5, n = (ci & 31) * 4;
      v4f x = *(const v4f*)&W[(size_t)(k0 + k) * N + n0 + n];
      Bs[(n + 0) * 64 + k] = (bf16)x[0];
      Bs[(n + 1) * 64 + k] = (bf16)x[1];
      Bs[(n + 2) * 64 + k] = (bf16)x[2];
      Bs[(n + 3) * 64 + k] = (bf16)x[3];
    }
    if (k0 + 64 < K) {  // prefetch next K-tile into cache
      __builtin_prefetch(&A[(size_t)(m0 + (tid >> 4)) * K + k0 + 64 + (tid & 15) * 4], 0, 1);
      __builtin_prefetch(&W[(size_t)(k0 + 64 + (tid >> 5)) * N + n0 + (tid & 31) * 4], 0, 1);
    }
    __syncthreads();
#pragma unroll
    for (int ks = 0; ks < 2; ++ks) {
      v16bf a0 = load_a_lds(&As[(wm)*64 + ks * 32], 64, lane);
      v16bf a1 = load_a_lds(&As[(wm + 16) * 64 + ks * 32], 64, lane);
#pragma unroll
      for (int nf = 0; nf < 4; ++nf) {
        v16bf bb = load_b_lds(&Bs[(wn + nf * 16) * 64 + ks * 32], 64, lane);
        acc[0][nf] = wmma_bf16(a0, bb, acc[0][nf]);
        acc[1][nf] = wmma_bf16(a1, bb, acc[1][nf]);
      }
    }
    __syncthreads();
  }

  const int half = (lane & 16) ? 8 : 0;
  const int ncol = lane & 15;
#pragma unroll
  for (int mf = 0; mf < 2; ++mf)
#pragma unroll
    for (int nf = 0; nf < 4; ++nf)
#pragma unroll
      for (int r = 0; r < 8; ++r) {
        int row = m0 + wm + mf * 16 + r + half;
        int col = n0 + wn + nf * 16 + ncol;
        float v = acc[mf][nf][r] + bias[col];
        if (mode == 1)
          v = 0.5f * v * (1.0f + erff(v * 0.70710678118f));
        else if (mode == 2)
          v += res[(size_t)row * N + col];
        C[(size_t)row * N + col] = v;
      }
}

// ---------------------------------------------------------------------------
// Sliding-window local attention (flash style) + folded-in global column.
// One block per (b, h, chunk). 256 threads = 8 waves, wave w -> 32 query rows.
// qkv layout: [B, S, 3*768] with q|k|v at col offsets 0/768/1536, head h at h*64.
// ---------------------------------------------------------------------------
__global__ __launch_bounds__(256) void attn_local(
    const float* __restrict__ qkv, const int* __restrict__ mask,
    float* __restrict__ out /* [B,S,768] */) {
  const int NC = 8;  // S / 256
  int b  = blockIdx.x / (12 * NC);
  int hc = blockIdx.x % (12 * NC);
  int h = hc / NC, c = hc % NC;

  __shared__ __align__(16) bf16 kt[64 * 64];      // [key][d]  (B for scores)
  __shared__ __align__(16) bf16 vt[64 * 64];      // [d][key]  (B for out)
  __shared__ __align__(16) bf16 pb[8][32 * 64];   // per-wave P tiles [m][key]
  __shared__ int   kval[64];
  __shared__ float k0f[64], v0f[64];

  const int tid = threadIdx.x, lane = tid & 31, wid = tid >> 5;
  const int half = (lane & 16) ? 8 : 0;
  const int ncol = lane & 15;
  const int wm0 = wid * 32;

  const float* qbase = qkv + (size_t)b * 2048 * 2304 + h * 64;
  const float* kbase = qkv + (size_t)b * 2048 * 2304 + 768 + h * 64;
  const float* vbase = qkv + (size_t)b * 2048 * 2304 + 1536 + h * 64;

  if (tid < 64) {
    k0f[tid] = kbase[tid];  // key/value at global position 0
    v0f[tid] = vbase[tid];
  }

  // q fragments are invariant across key tiles: hoist (scaled by 1/sqrt(D))
  v16bf qa[2][2];
#pragma unroll
  for (int mf = 0; mf < 2; ++mf)
#pragma unroll
    for (int ks = 0; ks < 2; ++ks)
      qa[mf][ks] = load_a_gf32(
          qbase + (size_t)(c * 256 + wm0 + mf * 16) * 2304 + ks * 32, 2304, lane,
          0.125f);

  float mr[2][8], lr[2][8];
  v8f oacc[2][4] = {};
#pragma unroll
  for (int mf = 0; mf < 2; ++mf)
#pragma unroll
    for (int r = 0; r < 8; ++r) { mr[mf][r] = -3.0e38f; lr[mf][r] = 0.f; }

  const int kpos0 = c * 256 - 256;

  for (int t = 0; t < 12; ++t) {
    __syncthreads();  // protect kt/vt from previous iteration readers
#pragma unroll
    for (int ci = tid; ci < 1024; ci += 256) {
      int kk = ci >> 4, d = (ci & 15) * 4;
      int kpos = kpos0 + t * 64 + kk;
      v4f kv = {0.f, 0.f, 0.f, 0.f}, vv = {0.f, 0.f, 0.f, 0.f};
      int valid = 0;
      if (kpos >= 0 && kpos < 2048) {
        kv = *(const v4f*)&kbase[(size_t)kpos * 2304 + d];
        vv = *(const v4f*)&vbase[(size_t)kpos * 2304 + d];
        valid = (mask[(size_t)b * 2048 + kpos] == 1) && (kpos != 0);
      }
      v4bf kp;
#pragma unroll
      for (int j = 0; j < 4; ++j) kp[j] = (bf16)kv[j];
      *(v4bf*)&kt[kk * 64 + d] = kp;
      vt[(d + 0) * 64 + kk] = (bf16)vv[0];
      vt[(d + 1) * 64 + kk] = (bf16)vv[1];
      vt[(d + 2) * 64 + kk] = (bf16)vv[2];
      vt[(d + 3) * 64 + kk] = (bf16)vv[3];
      if (d == 0) kval[kk] = valid;
    }
    __syncthreads();

    // scores: q(32x64) @ kt^T -> 32x64 tile
    v8f sc[2][4] = {};
#pragma unroll
    for (int ks = 0; ks < 2; ++ks)
#pragma unroll
      for (int nf = 0; nf < 4; ++nf) {
        v16bf bb = load_b_lds(&kt[(nf * 16) * 64 + ks * 32], 64, lane);
        sc[0][nf] = wmma_bf16(qa[0][ks], bb, sc[0][nf]);
        sc[1][nf] = wmma_bf16(qa[1][ks], bb, sc[1][nf]);
      }

    // mask + online softmax update
#pragma unroll
    for (int mf = 0; mf < 2; ++mf) {
      float tm[8], rs[8], scale[8];
#pragma unroll
      for (int r = 0; r < 8; ++r) {
        int m = wm0 + mf * 16 + r + half;  // chunk-local query index
        tm[r] = -3.0e38f;
#pragma unroll
        for (int nf = 0; nf < 4; ++nf) {
          int j = t * 64 + nf * 16 + ncol;  // 0..767 window col
          bool ok = (kval[nf * 16 + ncol] != 0) && (j >= m) && (j <= m + 512);
          float s = ok ? sc[mf][nf][r] : NEGV;
          sc[mf][nf][r] = s;
          tm[r] = fmaxf(tm[r], s);
        }
      }
#pragma unroll
      for (int o = 1; o < 16; o <<= 1)
#pragma unroll
        for (int r = 0; r < 8; ++r) tm[r] = fmaxf(tm[r], __shfl_xor(tm[r], o, 32));
#pragma unroll
      for (int r = 0; r < 8; ++r) {
        float nm = fmaxf(mr[mf][r], tm[r]);
        scale[r] = __expf(mr[mf][r] - nm);
        mr[mf][r] = nm;
        rs[r] = 0.f;
#pragma unroll
        for (int nf = 0; nf < 4; ++nf) {
          float p = sc[mf][nf][r];
          p = (p < -1.0e8f) ? 0.f : __expf(p - nm);
          sc[mf][nf][r] = p;
          rs[r] += p;
        }
      }
#pragma unroll
      for (int o = 1; o < 16; o <<= 1)
#pragma unroll
        for (int r = 0; r < 8; ++r) rs[r] += __shfl_xor(rs[r], o, 32);
#pragma unroll
      for (int r = 0; r < 8; ++r) lr[mf][r] = lr[mf][r] * scale[r] + rs[r];
#pragma unroll
      for (int nf = 0; nf < 4; ++nf)
#pragma unroll
        for (int r = 0; r < 8; ++r) oacc[mf][nf][r] *= scale[r];
      // store P (C layout -> LDS) for reload as WMMA A operand
#pragma unroll
      for (int nf = 0; nf < 4; ++nf)
#pragma unroll
        for (int r = 0; r < 8; ++r)
          pb[wid][(mf * 16 + r + half) * 64 + nf * 16 + ncol] =
              (bf16)sc[mf][nf][r];
    }

    // out += P(32x64) @ V(64x64); DS ops of the same wave are in-order
#pragma unroll
    for (int ks = 0; ks < 2; ++ks) {
      v16bf pa0 = load_a_lds(&pb[wid][ks * 32], 64, lane);
      v16bf pa1 = load_a_lds(&pb[wid][16 * 64 + ks * 32], 64, lane);
#pragma unroll
      for (int nf = 0; nf < 4; ++nf) {
        v16bf vb = load_b_lds(&vt[(nf * 16) * 64 + ks * 32], 64, lane);
        oacc[0][nf] = wmma_bf16(pa0, vb, oacc[0][nf]);
        oacc[1][nf] = wmma_bf16(pa1, vb, oacc[1][nf]);
      }
    }
  }

  // fold in the global column (key at position 0)
  const int m0ok = (mask[(size_t)b * 2048] == 1);
#pragma unroll
  for (int mf = 0; mf < 2; ++mf) {
    float g[8];
#pragma unroll
    for (int r = 0; r < 8; ++r) {
      int m = wm0 + mf * 16 + r + half;
      const float* qrow = qbase + (size_t)(c * 256 + m) * 2304;
      float part = 0.f;
#pragma unroll
      for (int dd = 0; dd < 4; ++dd) {
        int d = ncol + dd * 16;
        part += qrow[d] * 0.125f * k0f[d];
      }
      g[r] = part;
    }
#pragma unroll
    for (int o = 1; o < 16; o <<= 1)
#pragma unroll
      for (int r = 0; r < 8; ++r) g[r] += __shfl_xor(g[r], o, 32);
    float gsc[8], gpg[8];
#pragma unroll
    for (int r = 0; r < 8; ++r) {
      float gv = m0ok ? g[r] : NEGV;
      float nm = fmaxf(mr[mf][r], gv);
      gsc[r] = __expf(mr[mf][r] - nm);
      gpg[r] = (gv < -1.0e8f) ? 0.f : __expf(gv - nm);
      mr[mf][r] = nm;
      lr[mf][r] = lr[mf][r] * gsc[r] + gpg[r];
    }
#pragma unroll
    for (int nf = 0; nf < 4; ++nf)
#pragma unroll
      for (int r = 0; r < 8; ++r)
        oacc[mf][nf][r] = oacc[mf][nf][r] * gsc[r] + gpg[r] * v0f[nf * 16 + ncol];
  }

  // normalize + store
#pragma unroll
  for (int mf = 0; mf < 2; ++mf)
#pragma unroll
    for (int nf = 0; nf < 4; ++nf)
#pragma unroll
      for (int r = 0; r < 8; ++r) {
        int m = wm0 + mf * 16 + r + half;
        int d = nf * 16 + ncol;
        float inv = 1.0f / fmaxf(lr[mf][r], 1e-20f);
        out[((size_t)b * 2048 + c * 256 + m) * 768 + h * 64 + d] =
            oacc[mf][nf][r] * inv;
      }
}

// ---------------------------------------------------------------------------
// Global attention for token 0 (overwrites attn output row 0 per head)
// ---------------------------------------------------------------------------
__global__ __launch_bounds__(256) void attn_global(
    const float* __restrict__ gqkv, const int* __restrict__ mask,
    float* __restrict__ out) {
  int b = blockIdx.x / 12, h = blockIdx.x % 12;
  __shared__ float q0[64];
  __shared__ float gs[2048];
  __shared__ float red[256];
  const int tid = threadIdx.x;
  const float* base = gqkv + (size_t)b * 2048 * 2304;
  if (tid < 64) q0[tid] = base[h * 64 + tid] * 0.125f;
  __syncthreads();

  float lm = -3.0e38f;
  for (int s = tid; s < 2048; s += 256) {
    const float* kr = base + (size_t)s * 2304 + 768 + h * 64;
    float acc = 0.f;
#pragma unroll
    for (int d = 0; d < 64; d += 4) {
      v4f kv = *(const v4f*)(kr + d);
      acc += q0[d] * kv[0] + q0[d + 1] * kv[1] + q0[d + 2] * kv[2] +
             q0[d + 3] * kv[3];
    }
    if (mask[(size_t)b * 2048 + s] != 1) acc = NEGV;
    gs[s] = acc;
    lm = fmaxf(lm, acc);
  }
  red[tid] = lm; __syncthreads();
  for (int o = 128; o > 0; o >>= 1) {
    if (tid < o) red[tid] = fmaxf(red[tid], red[tid + o]);
    __syncthreads();
  }
  float mx = red[0]; __syncthreads();
  float ls = 0.f;
  for (int s = tid; s < 2048; s += 256) {
    float p = __expf(gs[s] - mx);
    gs[s] = p;
    ls += p;
  }
  red[tid] = ls; __syncthreads();
  for (int o = 128; o > 0; o >>= 1) {
    if (tid < o) red[tid] += red[tid + o];
    __syncthreads();
  }
  float l = fmaxf(red[0], 1e-20f); __syncthreads();

  int d = tid & 63, grp = tid >> 6;
  float acc = 0.f;
  for (int s = grp * 512; s < grp * 512 + 512; ++s)
    acc += gs[s] * base[(size_t)s * 2304 + 1536 + h * 64 + d];
  red[tid] = acc; __syncthreads();
  if (tid < 64) {
    float t2 = red[tid] + red[tid + 64] + red[tid + 128] + red[tid + 192];
    out[(size_t)b * 2048 * 768 + h * 64 + d] = t2 / l;
  }
}

// ---------------------------------------------------------------------------
// Row layernorm over E=768
// ---------------------------------------------------------------------------
__global__ __launch_bounds__(256) void ln_kernel(
    const float* __restrict__ in, const float* __restrict__ g,
    const float* __restrict__ bb, float* __restrict__ outp) {
  const int row = blockIdx.x, tid = threadIdx.x;
  __shared__ float red[256];
  const float* x = in + (size_t)row * 768;
  float s = 0.f;
  for (int i = tid; i < 768; i += 256) s += x[i];
  red[tid] = s; __syncthreads();
  for (int o = 128; o > 0; o >>= 1) { if (tid < o) red[tid] += red[tid + o]; __syncthreads(); }
  float mean = red[0] * (1.0f / 768.0f); __syncthreads();
  float v = 0.f;
  for (int i = tid; i < 768; i += 256) { float d = x[i] - mean; v += d * d; }
  red[tid] = v; __syncthreads();
  for (int o = 128; o > 0; o >>= 1) { if (tid < o) red[tid] += red[tid + o]; __syncthreads(); }
  float inv = rsqrtf(red[0] * (1.0f / 768.0f) + 1e-5f);
  for (int i = tid; i < 768; i += 256)
    outp[(size_t)row * 768 + i] = (x[i] - mean) * inv * g[i] + bb[i];
}

// ---------------------------------------------------------------------------
// pos_ids = cumsum(mask) * mask + 1  (per batch inclusive scan over S=2048)
// ---------------------------------------------------------------------------
__global__ __launch_bounds__(256) void poscum(const int* __restrict__ mask,
                                              int* __restrict__ pos) {
  const int b = blockIdx.x, tid = threadIdx.x;
  __shared__ int sb[256];
  int off = 0;
  for (int c0 = 0; c0 < 2048; c0 += 256) {
    int mv = mask[(size_t)b * 2048 + c0 + tid];
    int val = mv;
    sb[tid] = val; __syncthreads();
    for (int o = 1; o < 256; o <<= 1) {
      int add = (tid >= o) ? sb[tid - o] : 0;
      __syncthreads();
      val += add;
      sb[tid] = val;
      __syncthreads();
    }
    pos[(size_t)b * 2048 + c0 + tid] = (off + val) * mv + 1;
    off += sb[255];
    __syncthreads();
  }
}

// ---------------------------------------------------------------------------
// Embedding gather + layernorm
// ---------------------------------------------------------------------------
__global__ __launch_bounds__(256) void embed_ln(
    const int* __restrict__ ids, const int* __restrict__ tt,
    const int* __restrict__ pos, const float* __restrict__ we,
    const float* __restrict__ pe, const float* __restrict__ te,
    const float* __restrict__ g, const float* __restrict__ bb,
    float* __restrict__ outp) {
  const int tok = blockIdx.x, tid = threadIdx.x;
  __shared__ float xb[768];
  __shared__ float red[256];
  const int id = ids[tok], p = pos[tok], t = tt[tok];
  for (int i = tid; i < 768; i += 256)
    xb[i] = we[(size_t)id * 768 + i] + pe[(size_t)p * 768 + i] +
            te[(size_t)t * 768 + i];
  __syncthreads();
  float s = 0.f;
  for (int i = tid; i < 768; i += 256) s += xb[i];
  red[tid] = s; __syncthreads();
  for (int o = 128; o > 0; o >>= 1) { if (tid < o) red[tid] += red[tid + o]; __syncthreads(); }
  float mean = red[0] * (1.0f / 768.0f); __syncthreads();
  float v = 0.f;
  for (int i = tid; i < 768; i += 256) { float d = xb[i] - mean; v += d * d; }
  red[tid] = v; __syncthreads();
  for (int o = 128; o > 0; o >>= 1) { if (tid < o) red[tid] += red[tid + o]; __syncthreads(); }
  float inv = rsqrtf(red[0] * (1.0f / 768.0f) + 1e-5f);
  for (int i = tid; i < 768; i += 256)
    outp[(size_t)tok * 768 + i] = (xb[i] - mean) * inv * g[i] + bb[i];
}

// ---------------------------------------------------------------------------
// cls = tanh(x[:,0] @ cls_w + cls_b)   (M=2, tiny)
// ---------------------------------------------------------------------------
__global__ __launch_bounds__(256) void cls_kernel(
    const float* __restrict__ x, const float* __restrict__ W,
    const float* __restrict__ bvec, float* __restrict__ clsbuf) {
  const int b = blockIdx.x, tid = threadIdx.x;
  const float* row = x + (size_t)b * 2048 * 768;  // token 0 of batch b
  for (int e = tid; e < 768; e += 256) {
    float acc = bvec[e];
    for (int k = 0; k < 768; ++k) acc += row[k] * W[(size_t)k * 768 + e];
    clsbuf[(size_t)b * 768 + e] = tanhf(acc);
  }
}

__global__ __launch_bounds__(32) void out_kernel(
    const float* __restrict__ clsbuf, const float* __restrict__ W,
    const float* __restrict__ bvec, float* __restrict__ outp) {
  const int t = threadIdx.x;
  if (t < 18) {
    int b = t / 9, n = t % 9;
    float acc = bvec[n];
    for (int k = 0; k < 768; ++k)
      acc += clsbuf[(size_t)b * 768 + k] * W[(size_t)k * 9 + n];
    outp[t] = acc;
  }
}

// ---------------------------------------------------------------------------
// Driver
// ---------------------------------------------------------------------------
extern "C" void kernel_launch(void* const* d_in, const int* in_sizes, int n_in,
                              void* d_out, int out_size, void* d_ws, size_t ws_size,
                              hipStream_t stream) {
  (void)in_sizes; (void)n_in; (void)out_size; (void)ws_size;
  const int* ids      = (const int*)d_in[0];
  const int* mask     = (const int*)d_in[1];
  const int* tt       = (const int*)d_in[2];
  const float* we     = (const float*)d_in[3];
  const float* pe     = (const float*)d_in[4];
  const float* te     = (const float*)d_in[5];
  const float* lng    = (const float*)d_in[6];
  const float* lnb    = (const float*)d_in[7];
  const float* Wqkv   = (const float*)d_in[8];
  const float* bqkv   = (const float*)d_in[9];
  const float* Wg     = (const float*)d_in[10];
  const float* bg     = (const float*)d_in[11];
  const float* Wo     = (const float*)d_in[12];
  const float* bo     = (const float*)d_in[13];
  const float* ln1g   = (const float*)d_in[14];
  const float* ln1b   = (const float*)d_in[15];
  const float* Wi     = (const float*)d_in[16];
  const float* bi     = (const float*)d_in[17];
  const float* Wf     = (const float*)d_in[18];
  const float* bf_    = (const float*)d_in[19];
  const float* ln2g   = (const float*)d_in[20];
  const float* ln2b   = (const float*)d_in[21];
  const float* cls_w  = (const float*)d_in[22];
  const float* cls_b  = (const float*)d_in[23];
  const float* out_w  = (const float*)d_in[24];
  const float* out_b  = (const float*)d_in[25];

  // workspace layout (floats)
  float* ws   = (float*)d_ws;
  float* x    = ws;                      // 4096*768
  float* y    = x + 3145728;             // 4096*768
  float* a    = y + 3145728;             // 4096*768
  float* qkv  = a + 3145728;             // 4096*2304
  float* gq   = qkv + 9437184;           // 4096*2304
  float* hbuf = qkv;                     // 4096*3072, reuses qkv+gq region
  int*   posb = (int*)(gq + 9437184);    // 8192
  float* clsb = (float*)(posb + 8192);   // 1536

  poscum<<<2, 256, 0, stream>>>(mask, posb);
  embed_ln<<<4096, 256, 0, stream>>>(ids, tt, posb, we, pe, te, lng, lnb, x);

  for (int l = 0; l < 2; ++l) {
    const float* Wqkv_l = Wqkv + (size_t)l * 768 * 2304;
    const float* Wg_l   = Wg + (size_t)l * 768 * 2304;
    const float* Wo_l   = Wo + (size_t)l * 768 * 768;
    const float* Wi_l   = Wi + (size_t)l * 768 * 3072;
    const float* Wf_l   = Wf + (size_t)l * 3072 * 768;

    gemm_wmma<<<dim3(18, 32), 256, 0, stream>>>(x, Wqkv_l, bqkv + l * 2304,
                                                nullptr, qkv, 4096, 2304, 768, 0);
    gemm_wmma<<<dim3(18, 32), 256, 0, stream>>>(x, Wg_l, bg + l * 2304,
                                                nullptr, gq, 4096, 2304, 768, 0);
    attn_local<<<192, 256, 0, stream>>>(qkv, mask, a);
    attn_global<<<24, 256, 0, stream>>>(gq, mask, a);
    gemm_wmma<<<dim3(6, 32), 256, 0, stream>>>(a, Wo_l, bo + l * 768, x, y,
                                               4096, 768, 768, 2);
    ln_kernel<<<4096, 256, 0, stream>>>(y, ln1g + l * 768, ln1b + l * 768, x);
    gemm_wmma<<<dim3(24, 32), 256, 0, stream>>>(x, Wi_l, bi + l * 3072,
                                                nullptr, hbuf, 4096, 3072, 768, 1);
    gemm_wmma<<<dim3(6, 32), 256, 0, stream>>>(hbuf, Wf_l, bf_ + l * 768, x, y,
                                               4096, 768, 3072, 2);
    ln_kernel<<<4096, 256, 0, stream>>>(y, ln2g + l * 768, ln2b + l * 768, x);
  }

  cls_kernel<<<2, 256, 0, stream>>>(x, cls_w, cls_b, clsb);
  out_kernel<<<1, 32, 0, stream>>>(clsb, out_w, out_b, (float*)d_out);
}
